// FlashAttentionEngine_56710748176493
// MI455X (gfx1250) — matrix operand order repaired and checked by reading, not compiled, not run on hardware
//
#include <hip/hip_runtime.h>

#define S_LEN 2048
#define HID   1024
#define NH    16
#define HD    64
#define QTILE 128   // q rows per block (8 waves x 16 rows)
#define KTILE 32    // keys per LDS tile
#define NT    (S_LEN / KTILE)

// log2(e) folded into the Q pre-scale so p = exp2(score) directly;
// the uniform softmax scale cancels exactly in sum(p*V)/sum(p).
#define QSCALE (0.125f * 1.44269504f)

typedef __attribute__((ext_vector_type(16))) __bf16 v16bf;
typedef __attribute__((ext_vector_type(2)))  __bf16 v2bf;
typedef __attribute__((ext_vector_type(8)))  float  v8f;
typedef __attribute__((ext_vector_type(2)))  float  v2f;

struct U32x8 { uint4 lo; uint4 hi; };

// v16bf from two 16-byte LDS chunks (8 bf16 each) -> 2x ds_load_b128
static __device__ __forceinline__ v16bf ld2(const unsigned short* p0,
                                            const unsigned short* p1) {
  U32x8 u{*(const uint4*)p0, *(const uint4*)p1};
  return __builtin_bit_cast(v16bf, u);
}

static __device__ __forceinline__ unsigned short bfbits(float f) {
  return __builtin_bit_cast(unsigned short, (__bf16)f);
}
// fused pair conversion -> single v_cvt_pk_bf16_f32
static __device__ __forceinline__ unsigned int packbf(float a, float b) {
  v2f x{a, b};
  return __builtin_bit_cast(unsigned int, __builtin_convertvector(x, v2bf));
}

__global__ __launch_bounds__(256)
void fa_wmma_kernel(const float* __restrict__ Q,
                    const float* __restrict__ K,
                    const float* __restrict__ V,
                    float* __restrict__ O) {
  __shared__ __align__(16) unsigned short KtB[KTILE][72];  // [key][d]
  __shared__ __align__(16) unsigned short VtT[HD][40];     // [d][perm key]
  __shared__ __align__(16) unsigned short Ps[8][16][40];   // per-wave P

  const int tid  = threadIdx.x;
  const int wave = tid >> 5;
  const int lane = tid & 31;
  const int half = lane >> 4;
  const int l16  = lane & 15;

  const int qb = blockIdx.x;
  const int h  = blockIdx.y;
  const int b  = blockIdx.z;

  const size_t bh_off = (size_t)b * S_LEN * HID + (size_t)h * HD;
  const float* Qg = Q + bh_off;
  const float* Kg = K + bh_off;
  const float* Vg = V + bh_off;
  float*       Og = O + bh_off;

  // per-thread staging slots (2 float4 per matrix per tile)
  const int f0 = tid, f1 = tid + 256;
  const int r0 = f0 >> 4, c0_ = (f0 & 15) << 2;
  const int r1 = f1 >> 4, c1_ = (f1 & 15) << 2;
  const float* kp0 = Kg + (size_t)r0 * HID + c0_;
  const float* kp1 = Kg + (size_t)r1 * HID + c1_;
  const float* vp0 = Vg + (size_t)r0 * HID + c0_;
  const float* vp1 = Vg + (size_t)r1 * HID + c1_;
  unsigned int* kd0 = (unsigned int*)&KtB[r0][c0_];
  unsigned int* kd1 = (unsigned int*)&KtB[r1][c1_];
  const int pk0 = 2 * (r0 & 15) + (r0 >> 4);   // key permutation pi(k)=2*(k%16)+k/16
  const int pk1 = 2 * (r1 & 15) + (r1 >> 4);
  unsigned short* vd0 = &VtT[c0_][pk0];
  unsigned short* vd1 = &VtT[c1_][pk1];

  // ---- Q as two 16x32 bf16 A-fragments, pre-scaled by log2(e)/sqrt(64) ----
  const int qrow = qb * QTILE + wave * 16 + l16;
  v16bf aQ0, aQ1;
  {
    const float* qr = Qg + (size_t)qrow * HID;
    const int k0 = half * 8;
#pragma unroll
    for (int j = 0; j < 8; ++j) {
      aQ0[j]     = (__bf16)(qr[k0 + j]      * QSCALE);
      aQ0[j + 8] = (__bf16)(qr[16 + k0 + j] * QSCALE);
      aQ1[j]     = (__bf16)(qr[32 + k0 + j] * QSCALE);
      aQ1[j + 8] = (__bf16)(qr[48 + k0 + j] * QSCALE);
    }
  }

  // all-ones B: acc[4] accumulates P*1 = row sums (softmax denominator)
  v16bf bOne;
#pragma unroll
  for (int e = 0; e < 16; ++e)
    bOne[e] = __builtin_bit_cast(__bf16, (unsigned short)0x3F80);

  v8f acc[5];
#pragma unroll
  for (int t = 0; t < 5; ++t) acc[t] = (v8f){0,0,0,0,0,0,0,0};

  // prolog: prefetch tile 0
  float4 kv0 = *(const float4*)kp0;
  float4 kv1 = *(const float4*)kp1;
  float4 vv0 = *(const float4*)vp0;
  float4 vv1 = *(const float4*)vp1;

  for (int kt = 0; kt < NT; ++kt) {
    __syncthreads();                     // previous tile's LDS reads done
    // ---- store staged regs: K packed bf16, V transposed (perm keys) ----
    kd0[0] = packbf(kv0.x, kv0.y);  kd0[1] = packbf(kv0.z, kv0.w);
    kd1[0] = packbf(kv1.x, kv1.y);  kd1[1] = packbf(kv1.z, kv1.w);
    vd0[0]   = bfbits(vv0.x); vd0[40]  = bfbits(vv0.y);
    vd0[80]  = bfbits(vv0.z); vd0[120] = bfbits(vv0.w);
    vd1[0]   = bfbits(vv1.x); vd1[40]  = bfbits(vv1.y);
    vd1[80]  = bfbits(vv1.z); vd1[120] = bfbits(vv1.w);
    // ---- prefetch next tile (overlaps the 9 WMMAs below) ----
    if (kt + 1 < NT) {
      size_t adv = (size_t)(kt + 1) * KTILE * HID;
      kv0 = *(const float4*)(kp0 + adv);
      kv1 = *(const float4*)(kp1 + adv);
      vv0 = *(const float4*)(vp0 + adv);
      vv1 = *(const float4*)(vp1 + adv);
    }
    __syncthreads();                     // tile staged

    // ---- scores: S = Q*K^T (log2-domain), two 16x16 tiles ----
    v8f s0 = (v8f){0,0,0,0,0,0,0,0}, s1 = (v8f){0,0,0,0,0,0,0,0};
#pragma unroll
    for (int ck = 0; ck < 2; ++ck) {     // head-dim chunks 0..31 / 32..63
      const int doff = ck * 32 + half * 8;
      v16bf bk0 = ld2(&KtB[l16][doff],      &KtB[l16][doff + 16]);
      v16bf bk1 = ld2(&KtB[l16 + 16][doff], &KtB[l16 + 16][doff + 16]);
      const v16bf a = ck ? aQ1 : aQ0;
      s0 = __builtin_amdgcn_wmma_f32_16x16x32_bf16(false, a, false, bk0, (short)0, s0, false, false);
      s1 = __builtin_amdgcn_wmma_f32_16x16x32_bf16(false, a, false, bk1, (short)0, s1, false, false);
    }

    // ---- p = exp2(s), packed bf16 pair -> one b32 store per row ----
#pragma unroll
    for (int r = 0; r < 8; ++r) {
      float e0 = __builtin_amdgcn_exp2f(s0[r]);
      float e1 = __builtin_amdgcn_exp2f(s1[r]);
      *(unsigned int*)&Ps[wave][r + half * 8][2 * l16] = packbf(e0, e1);
    }
    // A fragment of P (16x32 over permuted key index): two b128 loads
    v16bf aP = ld2(&Ps[wave][l16][half * 8], &Ps[wave][l16][16 + half * 8]);

    // ---- O += P x V (+ ones column for the denominator), 5 WMMAs ----
#pragma unroll
    for (int t = 0; t < 4; ++t) {
      const int dcol = t * 16 + l16;
      v16bf bv = ld2(&VtT[dcol][half * 8], &VtT[dcol][16 + half * 8]);
      acc[t] = __builtin_amdgcn_wmma_f32_16x16x32_bf16(false, aP, false, bv, (short)0, acc[t], false, false);
    }
    acc[4] = __builtin_amdgcn_wmma_f32_16x16x32_bf16(false, aP, false, bOne, (short)0, acc[4], false, false);
  }

  // ---- epilogue: normalize by l (replicated across lanes) and store ----
#pragma unroll
  for (int r = 0; r < 8; ++r) {
    float inv = 1.f / acc[4][r];
    int row = qb * QTILE + wave * 16 + r + half * 8;
#pragma unroll
    for (int t = 0; t < 4; ++t)
      Og[(size_t)row * HID + t * 16 + l16] = acc[t][r] * inv;
  }
}

extern "C" void kernel_launch(void* const* d_in, const int* in_sizes, int n_in,
                              void* d_out, int out_size, void* d_ws, size_t ws_size,
                              hipStream_t stream) {
  const float* Q = (const float*)d_in[0];
  const float* K = (const float*)d_in[1];
  const float* V = (const float*)d_in[2];
  float* O = (float*)d_out;
  const int B = in_sizes[0] / (S_LEN * HID);
  dim3 grid(S_LEN / QTILE, NH, B);
  fa_wmma_kernel<<<grid, 256, 0, stream>>>(Q, K, V, O);
}
